// SimpleOldSparseCNN_18829136626386
// MI455X (gfx1250) — matrix operand order repaired and checked by reading, not compiled, not run on hardware
//
#include <hip/hip_runtime.h>
#include <math.h>

// ---------------- problem constants ----------------
#define BATCH   128
#define HH      128
#define WW      128
#define SZ      127            // conv output spatial dim
#define FIN     16129          // SZ*SZ
#define KP      16160          // FIN padded to multiple of 32 (505 * 32)
#define KSTEPS  505            // KP / 32
#define OUT3    256
#define OUTALL  768
#define SPLITK  16
#define STEPS_PER_SPLIT 32     // ceil(505/16)

typedef __attribute__((ext_vector_type(8)))  _Float16 v8h;
typedef __attribute__((ext_vector_type(16))) _Float16 v16h;
typedef __attribute__((ext_vector_type(8)))  float    v8f;

// ---------------- workspace layout (bytes) ----------------
// Ah : f16 A matrices, 3 * 128 * KP halves            = 12,410,880 B  @ 0
// Wh : f16 weights,    3 * 256 * KP halves            = 24,821,760 B  @ 12,410,880
// C  : f32 partial out, 128 * 768 floats              =    393,216 B  @ 37,232,640
#define AH_OFF 0
#define WH_OFF 12410880
#define C_OFF  37232640

// ---------------- zero C ----------------
__global__ void zeroC_kernel(float* __restrict__ C) {
    int t = blockIdx.x * blockDim.x + threadIdx.x;
    if (t < BATCH * OUTALL) C[t] = 0.0f;
}

// ---------------- conv(2x2) + tanh -> f16 A (zero padded K tail) ----------------
__global__ void conv_tanh_kernel(const float* __restrict__ x,
                                 const float* __restrict__ wr,
                                 const float* __restrict__ wg,
                                 const float* __restrict__ wb,
                                 _Float16* __restrict__ Ah) {
    size_t t = (size_t)blockIdx.x * blockDim.x + threadIdx.x;
    if (t >= (size_t)3 * BATCH * KP) return;
    int col  = (int)(t % KP);
    size_t rb = t / KP;            // 0 .. 3*128-1
    int b = (int)(rb % BATCH);
    int c = (int)(rb / BATCH);
    _Float16 out = (_Float16)0.0f;
    if (col < FIN) {
        int i = col / SZ;
        int j = col % SZ;
        const float* w = (c == 0) ? wr : (c == 1) ? wg : wb;
        const float* xp = x + (((size_t)c * BATCH + b) * HH + i) * WW + j;
        float v = xp[0] * w[0] + xp[1] * w[1] + xp[WW] * w[2] + xp[WW + 1] * w[3];
        out = (_Float16)tanhf(v);
    }
    Ah[t] = out;
}

// ---------------- fcw f32 -> f16 (zero padded K tail) ----------------
__global__ void wconv_kernel(const float* __restrict__ fr,
                             const float* __restrict__ fg,
                             const float* __restrict__ fb,
                             _Float16* __restrict__ Wh) {
    size_t t = (size_t)blockIdx.x * blockDim.x + threadIdx.x;
    if (t >= (size_t)3 * OUT3 * KP) return;
    int k  = (int)(t % KP);
    size_t ro = t / KP;            // 0 .. 3*256-1
    int o = (int)(ro % OUT3);
    int c = (int)(ro / OUT3);
    _Float16 out = (_Float16)0.0f;
    if (k < FIN) {
        const float* f = (c == 0) ? fr : (c == 1) ? fg : fb;
        out = (_Float16)f[(size_t)o * FIN + k];
    }
    Wh[t] = out;
}

// ---------------- fragment loaders (per ISA 16-bit layouts) ----------------
// A 16x32 frag: lane(0..15)=row r, K chunks [0..7] & [16..23]; lanes 16..31: K [8..15] & [24..31]
__device__ inline v16h load_fragA(const _Float16* __restrict__ p) {
    v8h lo = *(const v8h*)(p);
    v8h hi = *(const v8h*)(p + 16);
    return __builtin_shufflevector(lo, hi, 0,1,2,3,4,5,6,7,8,9,10,11,12,13,14,15);
}
// B 32x16 frag: lane(0..15)=col n, K [0..7] & [16..23]; lanes 16..31: K [8..15] & [24..31]
__device__ inline v16h load_fragB(const _Float16* __restrict__ p) {
    v8h lo = *(const v8h*)(p);
    v8h hi = *(const v8h*)(p + 16);
    return __builtin_shufflevector(lo, hi, 0,1,2,3,4,5,6,7,8,9,10,11,12,13,14,15);
}

__device__ inline v8f wmma_f16(v16h a, v16h b, v8f c) {
    return __builtin_amdgcn_wmma_f32_16x16x32_f16(false, a, false, b, (short)0, c, false, false);
}

// ---------------- split-K WMMA GEMM: C += A(128xKP) * W^T(KPx256) ----------------
// grid.x = 12 (channel*4 + ntile), grid.y = SPLITK, block = 256 (8 waves)
// wave(0..7): wm = wave&3 -> 32 M-rows; wn = wave>>2 -> 32 N-cols of the 64-wide tile
__global__ __launch_bounds__(256) void gemm_wmma_kernel(const _Float16* __restrict__ Ah,
                                                        const _Float16* __restrict__ Wh,
                                                        float* __restrict__ C) {
    const int chan  = blockIdx.x >> 2;
    const int ntile = blockIdx.x & 3;
    int ks0 = blockIdx.y * STEPS_PER_SPLIT;
    int ks1 = ks0 + STEPS_PER_SPLIT;
    if (ks1 > KSTEPS) ks1 = KSTEPS;

    const int wave   = threadIdx.x >> 5;
    const int lane   = threadIdx.x & 31;
    const int laneLo = lane & 15;
    const int laneHi = lane >> 4;
    const int m0 = (wave & 3) * 32;                 // wave's M base (0..96)
    const int n0 = ntile * 64 + (wave >> 2) * 32;   // wave's N base within channel (0..224)

    const _Float16* Abase = Ah + (size_t)chan * BATCH * KP;
    const _Float16* Wbase = Wh + (size_t)chan * OUT3 * KP;

    // per-lane row pointers (row-major, stride KP halves; all 16B aligned)
    const _Float16* a0p = Abase + (size_t)(m0 + laneLo) * KP + laneHi * 8;
    const _Float16* a1p = a0p + (size_t)16 * KP;
    const _Float16* b0p = Wbase + (size_t)(n0 + laneLo) * KP + laneHi * 8;
    const _Float16* b1p = b0p + (size_t)16 * KP;

    v8f acc00 = {}, acc01 = {}, acc10 = {}, acc11 = {};

    #pragma unroll 2
    for (int ks = ks0; ks < ks1; ++ks) {
        const int kb = ks * 32;
        v16h a0 = load_fragA(a0p + kb);
        v16h a1 = load_fragA(a1p + kb);
        v16h b0 = load_fragB(b0p + kb);
        v16h b1 = load_fragB(b1p + kb);
        acc00 = wmma_f16(a0, b0, acc00);
        acc01 = wmma_f16(a0, b1, acc01);
        acc10 = wmma_f16(a1, b0, acc10);
        acc11 = wmma_f16(a1, b1, acc11);
    }

    // C/D f32 16x16 layout: VGPR r, lanes 0-15 -> M=r, N=lane; lanes 16-31 -> M=r+8, N=lane-16
    const int rbase = m0 + laneHi * 8;
    const int cbase = chan * OUT3 + n0 + laneLo;
    #pragma unroll
    for (int r = 0; r < 8; ++r) {
        atomicAdd(&C[(size_t)(rbase + r)      * OUTALL + cbase],      acc00[r]);
        atomicAdd(&C[(size_t)(rbase + r)      * OUTALL + cbase + 16], acc01[r]);
        atomicAdd(&C[(size_t)(rbase + r + 16) * OUTALL + cbase],      acc10[r]);
        atomicAdd(&C[(size_t)(rbase + r + 16) * OUTALL + cbase + 16], acc11[r]);
    }
}

// ---------------- epilogue: out = tanh(C + bias) ----------------
__global__ void epilogue_kernel(const float* __restrict__ C,
                                const float* __restrict__ br,
                                const float* __restrict__ bg,
                                const float* __restrict__ bb,
                                float* __restrict__ out) {
    int t = blockIdx.x * blockDim.x + threadIdx.x;
    if (t >= BATCH * OUTALL) return;
    int col = t % OUTALL;
    int c = col / OUT3;
    int o = col % OUT3;
    const float* bias = (c == 0) ? br : (c == 1) ? bg : bb;
    out[t] = tanhf(C[t] + bias[o]);
}

extern "C" void kernel_launch(void* const* d_in, const int* in_sizes, int n_in,
                              void* d_out, int out_size, void* d_ws, size_t ws_size,
                              hipStream_t stream) {
    (void)in_sizes; (void)n_in; (void)out_size; (void)ws_size;
    const float* x   = (const float*)d_in[0];
    const float* wr  = (const float*)d_in[1];
    const float* wg  = (const float*)d_in[2];
    const float* wb  = (const float*)d_in[3];
    const float* fwr = (const float*)d_in[4];
    const float* fbr = (const float*)d_in[5];
    const float* fwg = (const float*)d_in[6];
    const float* fbg = (const float*)d_in[7];
    const float* fwb = (const float*)d_in[8];
    const float* fbb = (const float*)d_in[9];
    float* out = (float*)d_out;

    char* ws = (char*)d_ws;
    _Float16* Ah = (_Float16*)(ws + AH_OFF);
    _Float16* Wh = (_Float16*)(ws + WH_OFF);
    float*    C  = (float*)(ws + C_OFF);

    // 1) zero the f32 accumulator
    zeroC_kernel<<<(BATCH * OUTALL + 255) / 256, 256, 0, stream>>>(C);

    // 2) conv + tanh -> f16 A (zero-padded K)
    {
        size_t n = (size_t)3 * BATCH * KP;
        conv_tanh_kernel<<<(unsigned)((n + 255) / 256), 256, 0, stream>>>(x, wr, wg, wb, Ah);
    }

    // 3) fcw -> f16 (zero-padded K)
    {
        size_t n = (size_t)3 * OUT3 * KP;
        wconv_kernel<<<(unsigned)((n + 255) / 256), 256, 0, stream>>>(fwr, fwg, fwb, Wh);
    }

    // 4) split-K WMMA GEMM with f32 atomic reduction
    {
        dim3 grid(12, SPLITK);
        gemm_wmma_kernel<<<grid, 256, 0, stream>>>(Ah, Wh, C);
    }

    // 5) bias + tanh epilogue
    epilogue_kernel<<<(BATCH * OUTALL + 255) / 256, 256, 0, stream>>>(C, fbr, fbg, fbb, out);
}
